// EquivariantGraphConvolutionalLayer_73873437491456
// MI455X (gfx1250) — matrix operand order repaired and checked by reading, not compiled, not run on hardware
//
#include <hip/hip_runtime.h>
#include <hip/hip_fp16.h>

typedef _Float16 half_t;
typedef __attribute__((ext_vector_type(16))) _Float16 v16h;
typedef __attribute__((ext_vector_type(8)))  float    v8f;

#define NF 64

// fragment table offsets inside d_ws (1 fragment = 512 halves = 1KB)
#define OFF_EW1 0    // 20 frags: K padded to 160 (actual 129), N=64
#define OFF_EW2 20   // 8 frags : K=64
#define OFF_AW1 28   // 16 frags: K=128
#define OFF_NW1 44   // 16 frags: K=128
#define OFF_NW2 60   // 8 frags : K=64
#define OFF_CW1 68   // 8 frags : K=64
#define N_FRAGS 76

// CDNA5 hardware tanh (TRANS32 pipe, co-executes with XDL/WMMA).
// Trailing v_nop covers the "1 independent op before TRANS result use"
// hazard rule, since inline asm bypasses compiler hazard insertion.
__device__ __forceinline__ float fast_tanh(float x) {
  float y;
  asm volatile("v_tanh_f32 %0, %1\n\tv_nop" : "=v"(y) : "v"(x));
  return y;
}
__device__ __forceinline__ float tanhshrink_f(float x) { return x - fast_tanh(x); }
__device__ __forceinline__ float sigmoid_f(float x) {
  return 0.5f * (1.f + fast_tanh(0.5f * x));
}

// Build an A-matrix fragment (16x32 f16) for this lane from an LDS row.
// ISA layout: lane holds row M=lane%16; halves j=0..7 -> K = kbase + 8*hi + j,
// halves j=8..15 -> K = kbase + 16 + 8*hi + (j-8).  Two 16B LDS loads.
__device__ __forceinline__ v16h lds_a_frag(const half_t* rowp, int kbase, int hi) {
  union { v16h h; uint4 q[2]; } u;
  u.q[0] = *reinterpret_cast<const uint4*>(rowp + kbase + 8 * hi);
  u.q[1] = *reinterpret_cast<const uint4*>(rowp + kbase + 16 + 8 * hi);
  return u.h;
}

// Load a prepacked B-matrix fragment (32x16 f16) from the workspace.
__device__ __forceinline__ v16h w_frag(const half_t* __restrict__ wf, int f, int lane) {
  union { v16h h; uint4 q[2]; } u;
  const uint4* p = reinterpret_cast<const uint4*>(wf + ((size_t)f * 32 + lane) * 16);
  u.q[0] = p[0];
  u.q[1] = p[1];
  return u.h;
}

#define WMMA(a, b, c) \
  __builtin_amdgcn_wmma_f32_16x16x32_f16(false, (a), false, (b), (short)0, (c), false, false)

__device__ __forceinline__ void fatomic_add(float* p, float v) {
  __hip_atomic_fetch_add(p, v, __ATOMIC_RELAXED, __HIP_MEMORY_SCOPE_AGENT);
}

// ---------------------------------------------------------------------------
// Kernel 1: pack f32 weights [K][64] into f16 WMMA B fragments.
// B layout: lane holds column N = lane%16; halves j<8 -> K = 8*(lane>=16) + j, j>=8 -> K = 16 + 8*(lane>=16) + (j-8).
// ---------------------------------------------------------------------------
__global__ void egnn_pack_weights(const float* __restrict__ eW1, const float* __restrict__ eW2,
                                  const float* __restrict__ aW1, const float* __restrict__ nW1,
                                  const float* __restrict__ nW2, const float* __restrict__ cW1,
                                  half_t* __restrict__ wf) {
  int f = blockIdx.x;
  int lane = threadIdx.x;
  const float* W;
  int Ka, fl;
  if (f < 20)      { W = eW1; Ka = 129; fl = f; }
  else if (f < 28) { W = eW2; Ka = 64;  fl = f - 20; }
  else if (f < 44) { W = aW1; Ka = 128; fl = f - 28; }
  else if (f < 60) { W = nW1; Ka = 128; fl = f - 44; }
  else if (f < 68) { W = nW2; Ka = 64;  fl = f - 60; }
  else             { W = cW1; Ka = 64;  fl = f - 68; }
  int kt = fl >> 2, nt = fl & 3;
  int n = nt * 16 + (lane & 15);
  int hi = lane >> 4;
  half_t* dst = wf + ((size_t)f * 32 + lane) * 16;
#pragma unroll
  for (int j = 0; j < 16; ++j) {
    int k = kt * 32 + ((j < 8) ? (8 * hi + j) : (16 + 8 * hi + (j - 8)));
    float v = (k < Ka) ? W[(size_t)k * NF + n] : 0.f;
    dst[j] = (half_t)v;
  }
}

// ---------------------------------------------------------------------------
// Kernel 2: init outputs. agg accumulator lives in the nodes_out region of
// d_out (zeroed here); coord_out starts as coord so edge atomics add onto it.
// ---------------------------------------------------------------------------
__global__ void egnn_init_kernel(float* __restrict__ out_nodes, float* __restrict__ out_coord,
                                 const float* __restrict__ coord, int NN) {
  int idx = blockIdx.x * blockDim.x + threadIdx.x;
  if (idx < NN * NF) out_nodes[idx] = 0.f;
  if (idx < NN * 3)  out_coord[idx] = coord[idx];
}

// ---------------------------------------------------------------------------
// Kernel 3: edge pipeline. 8 waves/block, 16 edges/wave, no block barriers
// (per-wave LDS slices; LDS ops are in-order within a wave).
// ---------------------------------------------------------------------------
__global__ __launch_bounds__(256) void egnn_edge_kernel(
    const float* __restrict__ nodes, const float* __restrict__ coord,
    const int* __restrict__ erow, const int* __restrict__ ecol,
    const half_t* __restrict__ wf,
    const float* __restrict__ eb1, const float* __restrict__ eb2,
    const float* __restrict__ ab1, const float* __restrict__ aW2, const float* __restrict__ ab2,
    const float* __restrict__ cb1, const float* __restrict__ cW2, const float* __restrict__ cb2,
    float* __restrict__ out_coord, float* __restrict__ agg, int NE) {
  __shared__ __align__(16) half_t S_in[128][160];  // [src(64)|tgt(64)|radial|pad] f16
  __shared__ __align__(16) half_t Sh[128][64];     // hidden staging f16
  __shared__ float Srij[128][3];
  __shared__ int   Srow[128];
  __shared__ int   Scol[128];
  __shared__ float Sa[128];

  const int lane = threadIdx.x & 31;
  const int wave = threadIdx.x >> 5;
  const int col  = lane & 15;
  const int hi   = lane >> 4;
  const int lb   = wave * 16;
  const int ebase = blockIdx.x * 128 + lb;
  if (ebase >= NE) return;  // wave-uniform

  // ---- indices, rij, radial ----
  if (lane < 16) {
    int e = ebase + lane;
    int r = erow[e], c = ecol[e];
    Srow[lb + lane] = r;
    Scol[lb + lane] = c;
    float rx = coord[(size_t)r * 3 + 0] - coord[(size_t)c * 3 + 0];
    float ry = coord[(size_t)r * 3 + 1] - coord[(size_t)c * 3 + 1];
    float rz = coord[(size_t)r * 3 + 2] - coord[(size_t)c * 3 + 2];
    Srij[lb + lane][0] = rx; Srij[lb + lane][1] = ry; Srij[lb + lane][2] = rz;
    float rad = rx * rx + ry * ry + rz * rz;
    half_t* p = &S_in[lb + lane][128];
    p[0] = (half_t)rad;
#pragma unroll
    for (int j = 1; j < 32; ++j) p[j] = (half_t)0.f;
  }

  // ---- gather src/tgt features -> f16 LDS (float4 loads) ----
#pragma unroll
  for (int i = 0; i < 8; ++i) {
    int linear = i * 32 + lane;
    int e = linear >> 4, q = linear & 15;
    const float4 vs = *reinterpret_cast<const float4*>(nodes + (size_t)Srow[lb + e] * NF + q * 4);
    half_t* ds = &S_in[lb + e][q * 4];
    ds[0] = (half_t)vs.x; ds[1] = (half_t)vs.y; ds[2] = (half_t)vs.z; ds[3] = (half_t)vs.w;
    const float4 vt = *reinterpret_cast<const float4*>(nodes + (size_t)Scol[lb + e] * NF + q * 4);
    half_t* dt = &S_in[lb + e][64 + q * 4];
    dt[0] = (half_t)vt.x; dt[1] = (half_t)vt.y; dt[2] = (half_t)vt.z; dt[3] = (half_t)vt.w;
  }

  const half_t* arow = &S_in[lb + col][0];
  const half_t* hrow = &Sh[lb + col][0];
  v8f zero = {};
  v8f acc[4];

  // ---- attention layer 1: [src|tgt](K=128) @ aW1 ----
  acc[0] = acc[1] = acc[2] = acc[3] = zero;
#pragma unroll
  for (int kt = 0; kt < 4; ++kt) {
    v16h a = lds_a_frag(arow, kt * 32, hi);
#pragma unroll
    for (int nt = 0; nt < 4; ++nt)
      acc[nt] = WMMA(a, w_frag(wf, OFF_AW1 + kt * 4 + nt, lane), acc[nt]);
  }
#pragma unroll
  for (int nt = 0; nt < 4; ++nt) {
    float b = ab1[nt * 16 + col];
#pragma unroll
    for (int r = 0; r < 8; ++r)
      Sh[lb + r + 8 * hi][nt * 16 + col] = (half_t)tanhshrink_f(acc[nt][r] + b);
  }
  // ---- sigmoid attention gate ----
  if (lane < 16) {
    float s = ab2[0];
    const half_t* hp = &Sh[lb + lane][0];
    for (int k = 0; k < NF; ++k) s += (float)hp[k] * aW2[k];
    Sa[lb + lane] = sigmoid_f(s);
  }

  // ---- edge MLP layer 1: [src|tgt|radial](K=160 padded) @ eW1 ----
  acc[0] = acc[1] = acc[2] = acc[3] = zero;
#pragma unroll
  for (int kt = 0; kt < 5; ++kt) {
    v16h a = lds_a_frag(arow, kt * 32, hi);
#pragma unroll
    for (int nt = 0; nt < 4; ++nt)
      acc[nt] = WMMA(a, w_frag(wf, OFF_EW1 + kt * 4 + nt, lane), acc[nt]);
  }
#pragma unroll
  for (int nt = 0; nt < 4; ++nt) {
    float b = eb1[nt * 16 + col];
#pragma unroll
    for (int r = 0; r < 8; ++r)
      Sh[lb + r + 8 * hi][nt * 16 + col] = (half_t)tanhshrink_f(acc[nt][r] + b);
  }

  // ---- edge MLP layer 2 + attention scale -> edge_feat in S_in[:,0:64] ----
  acc[0] = acc[1] = acc[2] = acc[3] = zero;
#pragma unroll
  for (int kt = 0; kt < 2; ++kt) {
    v16h a = lds_a_frag(hrow, kt * 32, hi);
#pragma unroll
    for (int nt = 0; nt < 4; ++nt)
      acc[nt] = WMMA(a, w_frag(wf, OFF_EW2 + kt * 4 + nt, lane), acc[nt]);
  }
  float av[8];
#pragma unroll
  for (int r = 0; r < 8; ++r) av[r] = Sa[lb + r + 8 * hi];
#pragma unroll
  for (int nt = 0; nt < 4; ++nt) {
    float b = eb2[nt * 16 + col];
#pragma unroll
    for (int r = 0; r < 8; ++r) {
      float x = tanhshrink_f(acc[nt][r] + b) * av[r];
      S_in[lb + r + 8 * hi][nt * 16 + col] = (half_t)x;  // edge_feat
    }
  }

  // ---- coord MLP layer 1: edge_feat @ cW1 ----
  acc[0] = acc[1] = acc[2] = acc[3] = zero;
#pragma unroll
  for (int kt = 0; kt < 2; ++kt) {
    v16h a = lds_a_frag(arow, kt * 32, hi);
#pragma unroll
    for (int nt = 0; nt < 4; ++nt)
      acc[nt] = WMMA(a, w_frag(wf, OFF_CW1 + kt * 4 + nt, lane), acc[nt]);
  }
#pragma unroll
  for (int nt = 0; nt < 4; ++nt) {
    float b = cb1[nt * 16 + col];
#pragma unroll
    for (int r = 0; r < 8; ++r)
      Sh[lb + r + 8 * hi][nt * 16 + col] = (half_t)tanhshrink_f(acc[nt][r] + b);
  }

  // ---- per-edge coord scalar + coord scatter (f32 global atomics) ----
  if (lane < 16) {
    float s = cb2[0];
    const half_t* cp = &Sh[lb + lane][0];
    for (int k = 0; k < NF; ++k) s += (float)cp[k] * cW2[k];
    int r = Srow[lb + lane];
    fatomic_add(&out_coord[(size_t)r * 3 + 0], Srij[lb + lane][0] * s);
    fatomic_add(&out_coord[(size_t)r * 3 + 1], Srij[lb + lane][1] * s);
    fatomic_add(&out_coord[(size_t)r * 3 + 2], Srij[lb + lane][2] * s);
  }

  // ---- edge_feat scatter into agg ----
#pragma unroll
  for (int i = 0; i < 32; ++i) {
    int linear = i * 32 + lane;
    int e = linear >> 6, k = linear & 63;
    fatomic_add(&agg[(size_t)Srow[lb + e] * NF + k], (float)S_in[lb + e][k]);
  }
}

// ---------------------------------------------------------------------------
// Kernel 4: node update. agg is read from out_nodes, then the same wave
// overwrites its rows (loads complete before WMMA consumes them -> before
// any store issues, so aliasing is safe).
// ---------------------------------------------------------------------------
__global__ __launch_bounds__(256) void egnn_node_kernel(
    const float* __restrict__ nodes, const half_t* __restrict__ wf,
    const float* __restrict__ nb1, const float* __restrict__ nb2,
    float* __restrict__ out_nodes, int NN) {
  __shared__ __align__(16) half_t Sm[128][64];
  const int lane = threadIdx.x & 31;
  const int wave = threadIdx.x >> 5;
  const int col = lane & 15, hi = lane >> 4, lb = wave * 16;
  int tile = (blockIdx.x * 8 + wave) * 16;
  if (tile >= NN) return;  // wave-uniform
  int row = tile + col;
  if (row >= NN) row = NN - 1;

  v8f zero = {};
  v8f acc[4];
  acc[0] = acc[1] = acc[2] = acc[3] = zero;
#pragma unroll
  for (int kt = 0; kt < 4; ++kt) {
    const float* src = (kt < 2) ? nodes : out_nodes;  // out_nodes holds agg
    int kb = (kt & 1) * 32;
    union { v16h h; half_t e[16]; } u;
    const float* p0 = src + (size_t)row * NF + kb + 8 * hi;
    const float* p1 = p0 + 16;
#pragma unroll
    for (int j = 0; j < 8; ++j) { u.e[j] = (half_t)p0[j]; u.e[8 + j] = (half_t)p1[j]; }
#pragma unroll
    for (int nt = 0; nt < 4; ++nt)
      acc[nt] = WMMA(u.h, w_frag(wf, OFF_NW1 + kt * 4 + nt, lane), acc[nt]);
  }
#pragma unroll
  for (int nt = 0; nt < 4; ++nt) {
    float b = nb1[nt * 16 + col];
#pragma unroll
    for (int r = 0; r < 8; ++r)
      Sm[lb + r + 8 * hi][nt * 16 + col] = (half_t)tanhshrink_f(acc[nt][r] + b);
  }

  acc[0] = acc[1] = acc[2] = acc[3] = zero;
  const half_t* mrow = &Sm[lb + col][0];
#pragma unroll
  for (int kt = 0; kt < 2; ++kt) {
    v16h a = lds_a_frag(mrow, kt * 32, hi);
#pragma unroll
    for (int nt = 0; nt < 4; ++nt)
      acc[nt] = WMMA(a, w_frag(wf, OFF_NW2 + kt * 4 + nt, lane), acc[nt]);
  }
#pragma unroll
  for (int nt = 0; nt < 4; ++nt) {
    float b = nb2[nt * 16 + col];
#pragma unroll
    for (int r = 0; r < 8; ++r) {
      int m = tile + r + 8 * hi;
      if (m < NN) {
        size_t idx = (size_t)m * NF + nt * 16 + col;
        out_nodes[idx] = nodes[idx] + acc[nt][r] + b;
      }
    }
  }
}

// ---------------------------------------------------------------------------
extern "C" void kernel_launch(void* const* d_in, const int* in_sizes, int n_in,
                              void* d_out, int out_size, void* d_ws, size_t ws_size,
                              hipStream_t stream) {
  const float* nodes = (const float*)d_in[0];
  const float* coord = (const float*)d_in[1];
  const int*   edges = (const int*)d_in[2];
  const float* eW1 = (const float*)d_in[3];   const float* eb1 = (const float*)d_in[4];
  const float* eW2 = (const float*)d_in[5];   const float* eb2 = (const float*)d_in[6];
  const float* aW1 = (const float*)d_in[7];   const float* ab1 = (const float*)d_in[8];
  const float* aW2 = (const float*)d_in[9];   const float* ab2 = (const float*)d_in[10];
  const float* nW1 = (const float*)d_in[11];  const float* nb1 = (const float*)d_in[12];
  const float* nW2 = (const float*)d_in[13];  const float* nb2 = (const float*)d_in[14];
  const float* cW1 = (const float*)d_in[15];  const float* cb1 = (const float*)d_in[16];
  const float* cW2 = (const float*)d_in[17];  const float* cb2 = (const float*)d_in[18];

  int NN = in_sizes[0] / NF;
  int NE = in_sizes[2] / 2;
  float* out_nodes = (float*)d_out;
  float* out_coord = out_nodes + (size_t)NN * NF;
  half_t* wf = (half_t*)d_ws;

  egnn_pack_weights<<<N_FRAGS, 32, 0, stream>>>(eW1, eW2, aW1, nW1, nW2, cW1, wf);

  int initN = NN * NF;
  egnn_init_kernel<<<(initN + 255) / 256, 256, 0, stream>>>(out_nodes, out_coord, coord, NN);

  egnn_edge_kernel<<<(NE + 127) / 128, 256, 0, stream>>>(
      nodes, coord, edges, edges + NE, wf,
      eb1, eb2, ab1, aW2, ab2, cb1, cW2, cb2, out_coord, out_nodes, NE);

  int ntiles = (NN + 15) / 16;
  egnn_node_kernel<<<(ntiles + 7) / 8, 256, 0, stream>>>(nodes, wf, nb1, nb2, out_nodes, NN);
}
